// FSEncoder_PSP_44779329028473
// MI455X (gfx1250) — matrix operand order repaired and checked by reading, not compiled, not run on hardware
//
#include <hip/hip_runtime.h>
#include <cstddef>

typedef __bf16 bf16;
typedef __attribute__((ext_vector_type(16))) __bf16 v16bf;
typedef __attribute__((ext_vector_type(8)))  __bf16 v8bf;
typedef __attribute__((ext_vector_type(8)))  float  v8f;
typedef __attribute__((ext_vector_type(4)))  int    v4i;

// CDNA5 async global->LDS copy path (ASYNCcnt), with safe fallback.
#if __has_builtin(__builtin_amdgcn_global_load_async_to_lds_b128) && \
    __has_builtin(__builtin_amdgcn_s_wait_asynccnt)
#define USE_ASYNC_LDS 1
#else
#define USE_ASYNC_LDS 0
#endif

__device__ __forceinline__ void async_b128(const bf16* g, bf16* l) {
#if USE_ASYNC_LDS
    __builtin_amdgcn_global_load_async_to_lds_b128(
        (__attribute__((address_space(1))) v4i*)g,
        (__attribute__((address_space(3))) v4i*)l, 0, 0);
#else
    *(v8bf*)l = *(const v8bf*)g;
#endif
}

__device__ __forceinline__ void async_wait() {
#if USE_ASYNC_LDS
    __builtin_amdgcn_s_wait_asynccnt(0);
#endif
}

// ---------------------------------------------------------------------------
// Implicit-GEMM conv via WMMA bf16 (D = W x im2col), fp32 accumulate.
// Block tile: 128(M=Cout) x 256(N=spatial), 8 waves (wave32), wave tile 64x64.
// Double-buffered LDS; weight tile staged with async global->LDS b128 copies.
// Weights buffer is zero-padded to Mpad rows so async staging needs no guards.
// ---------------------------------------------------------------------------
template <int KH, int KW, int PAD>
__global__ __launch_bounds__(256) void k_conv_wmma(
    const bf16* __restrict__ X,   // [B, Cin, H, W] bf16
    const bf16* __restrict__ Wt,  // [Mpad, Kpad]   bf16, zero-padded both dims
    float* __restrict__ Y,        // [B, Cout, Ho, Wo] fp32
    int Cin, int H, int W, int Cout, int Ho, int Wo, int stride, int Kpad)
{
    constexpr int KHW = KH * KW;
    __shared__ __align__(32) bf16 aS[2][128 * 32];   // [m][k]
    __shared__ __align__(32) bf16 bS[2][256 * 32];   // [n][k] (n-major -> contiguous frags)

    const int b    = blockIdx.z;
    const int n0   = blockIdx.x * 256;
    const int m0   = blockIdx.y * 128;
    const int tid  = threadIdx.x;
    const int lane = tid & 31;
    const int wave = tid >> 5;
    const int wm   = wave & 1;        // M offset wm*64
    const int wn   = wave >> 1;       // N offset wn*64
    const int Nsp  = Ho * Wo;
    const int Kraw = Cin * KHW;

    // per-thread im2col column (fixed for whole K loop): column n0+tid
    const int n_g = n0 + tid;
    const bool nValid = (n_g < Nsp);
    const int oh  = nValid ? (n_g / Wo) : 0;
    const int ow  = nValid ? (n_g - oh * Wo) : 0;
    const int ohs = oh * stride - PAD;
    const int ows = ow * stride - PAD;
    const size_t xbase = (size_t)b * Cin * H * W;

    v8f acc[4][4] = {};

    const int nk = Kpad >> 5;

    // ---- staging helpers ----
    auto stage = [&](int buf, int k0) {
        // A tile (weights): 128x32, 2 x b128 per thread, fully in-bounds (padded)
        #pragma unroll
        for (int it = 0; it < 2; ++it) {
            int idx = tid + it * 256;          // 0..511
            int row = idx >> 2;                // 0..127
            int seg = idx & 3;                 // 4 x 8 bf16 per row
            async_b128(Wt + (size_t)(m0 + row) * Kpad + k0 + seg * 8,
                       &aS[buf][row * 32 + seg * 8]);
        }
        // B tile (im2col): this thread's column, 32 k values
        #pragma unroll 4
        for (int j = 0; j < 32; ++j) {
            int k = k0 + j;
            bf16 val = (bf16)0.f;
            if (nValid && k < Kraw) {
                int ci = k / KHW;
                int rs = k - ci * KHW;
                int r  = rs / KW;
                int ss = rs - r * KW;
                int ih = ohs + r;
                int iw = ows + ss;
                if ((unsigned)ih < (unsigned)H && (unsigned)iw < (unsigned)W)
                    val = X[xbase + ((size_t)ci * H + ih) * W + iw];
            }
            bS[buf][tid * 32 + j] = val;
        }
    };

    stage(0, 0);
    async_wait();
    __syncthreads();

    for (int kt = 0; kt < nk; ++kt) {
        const int cur = kt & 1;
        if (kt + 1 < nk)
            stage(cur ^ 1, (kt + 1) << 5);

        // ---- fragments from buffer `cur` ----
        v16bf bfm[4];
        #pragma unroll
        for (int j = 0; j < 4; ++j) {
            const bf16* col = &bS[cur][((wn * 64 + j * 16 + (lane & 15)) * 32) + (lane >> 4) * 16];
            bfm[j] = *(const v16bf*)col;                 // one 32B ds load
        }
        v16bf afm[4];
        #pragma unroll
        for (int i = 0; i < 4; ++i) {
            const bf16* row = &aS[cur][((wm * 64 + i * 16 + (lane & 15)) * 32) + (lane >> 4) * 16];
            v8bf lo = *(const v8bf*)row;                 // k = h*16 .. h*16+7
            v8bf hi = *(const v8bf*)(row + 8);           // k = h*16+8 .. h*16+15
            afm[i] = __builtin_shufflevector(lo, hi, 0,1,2,3,4,5,6,7,8,9,10,11,12,13,14,15);
        }
        // ---- 16 WMMAs per wave per K step ----
        #pragma unroll
        for (int i = 0; i < 4; ++i)
            #pragma unroll
            for (int j = 0; j < 4; ++j)
                acc[i][j] = __builtin_amdgcn_wmma_f32_16x16x32_bf16(
                    false, afm[i], false, bfm[j], (short)0, acc[i][j], false, false);

        if (kt + 1 < nk)
            async_wait();
        __syncthreads();
    }

    // ---- store C/D: VGPR e -> M = e + (lane>>4)*8, N = lane&15 ----
    #pragma unroll
    for (int i = 0; i < 4; ++i) {
        int mBase = m0 + wm * 64 + i * 16 + (lane >> 4) * 8;
        #pragma unroll
        for (int j = 0; j < 4; ++j) {
            int nGl = n0 + wn * 64 + j * 16 + (lane & 15);
            if (nGl < Nsp) {
                #pragma unroll
                for (int e = 0; e < 8; ++e) {
                    int m = mBase + e;
                    if (m < Cout)
                        Y[((size_t)b * Cout + m) * (size_t)Nsp + nGl] = acc[i][j][e];
                }
            }
        }
    }
}

// ---------------------------------------------------------------------------
// Elementwise / reduction helper kernels
// ---------------------------------------------------------------------------
__global__ void k_cvt_bf16(const float* __restrict__ s, bf16* __restrict__ d, int n) {
    int i = blockIdx.x * 256 + threadIdx.x;
    if (i < n) d[i] = (bf16)s[i];
}

// zero-pads both K (to Kpad) and M (to Mpad) so conv A-staging is guard-free
__global__ void k_cvt_pad_w(const float* __restrict__ s, bf16* __restrict__ d,
                            int Cout, int Kraw, int Kpad, int n) {
    int i = blockIdx.x * 256 + threadIdx.x;
    if (i < n) {
        int m = i / Kpad, k = i - m * Kpad;
        d[i] = (m < Cout && k < Kraw) ? (bf16)s[(size_t)m * Kraw + k] : (bf16)0.f;
    }
}

__global__ __launch_bounds__(256) void k_inorm_stats(const float* __restrict__ X,
                                                     float* __restrict__ mean,
                                                     float* __restrict__ rstd, int HW) {
    __shared__ float sh[512];
    int bc = blockIdx.x, tid = threadIdx.x;
    const float* p = X + (size_t)bc * HW;
    float s = 0.f, s2 = 0.f;
    for (int i = tid; i < HW; i += 256) { float v = p[i]; s += v; s2 += v * v; }
    sh[tid] = s; sh[256 + tid] = s2;
    __syncthreads();
    for (int w = 128; w > 0; w >>= 1) {
        if (tid < w) { sh[tid] += sh[tid + w]; sh[256 + tid] += sh[256 + tid + w]; }
        __syncthreads();
    }
    if (tid == 0) {
        float m = sh[0] / HW;
        float var = sh[256] / HW - m * m;
        mean[bc] = m;
        rstd[bc] = rsqrtf(var + 1e-5f);
    }
}

__global__ void k_inorm_bf16(const float* __restrict__ x, const float* __restrict__ mean,
                             const float* __restrict__ rstd, bf16* __restrict__ y, int HW) {
    int bc = blockIdx.x; float m = mean[bc], rs = rstd[bc];
    size_t base = (size_t)bc * HW;
    for (int i = threadIdx.x; i < HW; i += 256)
        y[base + i] = (bf16)((x[base + i] - m) * rs);
}

__global__ void k_inorm_f32(const float* __restrict__ x, const float* __restrict__ mean,
                            const float* __restrict__ rstd, float* __restrict__ y, int HW) {
    int bc = blockIdx.x; float m = mean[bc], rs = rstd[bc];
    size_t base = (size_t)bc * HW;
    for (int i = threadIdx.x; i < HW; i += 256)
        y[base + i] = (x[base + i] - m) * rs;
}

__global__ void k_inorm_prelu(const float* __restrict__ x, const float* __restrict__ mean,
                              const float* __restrict__ rstd, const float* __restrict__ a,
                              float* __restrict__ y, int C, int HW) {
    int bc = blockIdx.x; float m = mean[bc], rs = rstd[bc], al = a[bc % C];
    size_t base = (size_t)bc * HW;
    for (int i = threadIdx.x; i < HW; i += 256) {
        float v = (x[base + i] - m) * rs;
        y[base + i] = v >= 0.f ? v : al * v;
    }
}

__global__ void k_prelu_bf16(const float* __restrict__ x, const float* __restrict__ a,
                             bf16* __restrict__ y, int C, int HW) {
    int bc = blockIdx.x; float al = a[bc % C];
    size_t base = (size_t)bc * HW;
    for (int i = threadIdx.x; i < HW; i += 256) {
        float v = x[base + i];
        y[base + i] = (bf16)(v >= 0.f ? v : al * v);
    }
}

// SE gate: s = mean(inorm(x)) == 0 exactly  =>  g = sigmoid(sw2 @ relu(sb1) + sb2)
__global__ void k_se_gate(const float* __restrict__ sb1, const float* __restrict__ sw2,
                          const float* __restrict__ sb2, float* __restrict__ g,
                          int dep, int hid) {
    __shared__ float h[32];
    int tid = threadIdx.x;
    if (tid < hid) h[tid] = fmaxf(sb1[tid], 0.f);
    __syncthreads();
    for (int c = tid; c < dep; c += blockDim.x) {
        float acc = sb2[c];
        for (int k = 0; k < hid; ++k) acc += sw2[c * hid + k] * h[k];
        g[c] = 1.f / (1.f + __expf(-acc));
    }
}

// y = inorm(t2)*g[c] + sc   (fused norm + SE scale + residual add)
__global__ void k_combine(const float* __restrict__ t2, const float* __restrict__ mean,
                          const float* __restrict__ rstd, const float* __restrict__ g,
                          const float* __restrict__ sc, float* __restrict__ y,
                          int C, int HW) {
    int bc = blockIdx.x;
    float m = mean[bc], rs = rstd[bc], gg = g[bc % C];
    size_t base = (size_t)bc * HW;
    for (int i = threadIdx.x; i < HW; i += 256)
        y[base + i] = (t2[base + i] - m) * rs * gg + sc[base + i];
}

__global__ void k_slice2(const float* __restrict__ x, float* __restrict__ y,
                         int H, int W, int Ho, int Wo, int n) {
    int i = blockIdx.x * 256 + threadIdx.x;
    if (i < n) {
        int ow = i % Wo; int t = i / Wo;
        int oh = t % Ho; int bc = t / Ho;
        y[i] = x[((size_t)bc * H + oh * 2) * W + ow * 2];
    }
}

// segmap-masked average pooling into codes[b][s][coff + c]
__global__ __launch_bounds__(256) void k_pool_seg(const float* __restrict__ feats,
                                                  const int* __restrict__ seg,
                                                  float* __restrict__ codes,
                                                  int C, int h, int w,
                                                  int S, int coff, int Ctot) {
    __shared__ float msk[64 * 64];
    __shared__ float red[256];
    int b = blockIdx.y, s = blockIdx.x, tid = threadIdx.x;
    int hw = h * w;
    float cpart = 0.f;
    for (int i = tid; i < hw; i += 256) {
        int yy = i / w, xx = i - yy * w;
        int ih = yy * 256 / h, iw = xx * 256 / w;
        float mv = seg[(((size_t)b * S + s) * 256 + ih) * 256 + iw] > 0 ? 1.f : 0.f;
        msk[i] = mv;
        cpart += mv;
    }
    red[tid] = cpart;
    __syncthreads();
    for (int wd = 128; wd > 0; wd >>= 1) {
        if (tid < wd) red[tid] += red[tid + wd];
        __syncthreads();
    }
    float cnt = red[0];
    for (int c = tid; c < C; c += 256) {
        const float* f = feats + ((size_t)b * C + c) * hw;
        float sum = 0.f;
        for (int i = 0; i < hw; ++i) sum += msk[i] * f[i];
        codes[((size_t)b * S + s) * Ctot + coff + c] = cnt > 0.f ? sum / fmaxf(cnt, 1.f) : 0.f;
    }
}

__global__ void k_zero(float* __restrict__ d, int n) {
    int i = blockIdx.x * 256 + threadIdx.x;
    if (i < n) d[i] = 0.f;
}

// ---------------------------------------------------------------------------
// Host orchestration
// ---------------------------------------------------------------------------
struct Cfg { int inc, dep, st; };

static inline int cdivi(int a, int b) { return (a + b - 1) / b; }
static inline int kpad32(int k) { return (k + 31) & ~31; }

extern "C" void kernel_launch(void* const* d_in, const int* in_sizes, int n_in,
                              void* d_out, int out_size, void* d_ws, size_t ws_size,
                              hipStream_t stream) {
    (void)in_sizes; (void)n_in; (void)out_size; (void)ws_size;

    static const Cfg cfg[24] = {
        {64,128,2},{128,128,1},{128,128,1},
        {128,256,2},{256,256,1},{256,256,1},{256,256,1},
        {256,512,2},{512,512,1},{512,512,1},{512,512,1},{512,512,1},{512,512,1},
        {512,512,1},{512,512,1},{512,512,1},{512,512,1},{512,512,1},{512,512,1},
        {512,512,1},{512,512,1},
        {512,512,2},{512,512,1},{512,512,1}
    };

    // ---- inputs ----
    int idx = 0;
    const float* x_in   = (const float*)d_in[idx++];
    const int*   segmap = (const int*)  d_in[idx++];
    const float* conv0w = (const float*)d_in[idx++];
    const float* a0     = (const float*)d_in[idx++];
    const float *w1[24], *ap[24], *w2[24], *sb1[24], *sw2[24], *sb2[24], *scw[24];
    for (int i = 0; i < 24; ++i) {
        w1[i]  = (const float*)d_in[idx++];
        ap[i]  = (const float*)d_in[idx++];
        w2[i]  = (const float*)d_in[idx++];
        idx++;                               // sw1 (unused: s == 0 exactly)
        sb1[i] = (const float*)d_in[idx++];
        sw2[i] = (const float*)d_in[idx++];
        sb2[i] = (const float*)d_in[idx++];
        scw[i] = (cfg[i].inc != cfg[i].dep) ? (const float*)d_in[idx++] : nullptr;
    }

    // ---- workspace bump allocator ----
    char* base = (char*)d_ws;
    size_t off = 0;
    auto alloc = [&](size_t bytes) -> void* {
        void* p = base + off;
        off += (bytes + 255) & ~(size_t)255;
        return p;
    };
    float* X0   = (float*)alloc(8388608ull  * 4);   // activations ping
    float* X1   = (float*)alloc(8388608ull  * 4);   // activations pong
    float* T1   = (float*)alloc(16777216ull * 4);   // conv1 output (max 2x128x256x256)
    float* T2   = (float*)alloc(4194304ull  * 4);   // conv2 output
    float* SC   = (float*)alloc(4194304ull  * 4);   // shortcut buffer
    bf16*  BFA  = (bf16*) alloc(16777216ull * 2);   // bf16 staging (r / prelu)
    bf16*  BFB  = (bf16*) alloc(8388608ull  * 2);   // bf16 staging (x for sc-conv)
    bf16*  W1B  = (bf16*) alloc(2359296ull  * 2);   // 512 x 4608 max
    bf16*  W2B  = (bf16*) alloc(2359296ull  * 2);
    bf16*  WSB  = (bf16*) alloc(2359296ull  * 2);
    float* MEAN  = (float*)alloc(1024 * 4);
    float* RSTD  = (float*)alloc(1024 * 4);
    float* MEAN2 = (float*)alloc(1024 * 4);
    float* RSTD2 = (float*)alloc(1024 * 4);
    float* G     = (float*)alloc(512 * 4);

    const int B = 2, S = 12, Ctot = 1280;
    float* codes = (float*)d_out;                     // [2][12][1280]
    float* ztail = (float*)d_out + B * S * Ctot;      // zeros_like(x): 2*512*16*16

    auto cvtw = [&](const float* src, bf16* dst, int Cout, int Kraw, int Kp) {
        int Mpad = cdivi(Cout, 128) * 128;
        int n = Mpad * Kp;
        k_cvt_pad_w<<<cdivi(n, 256), 256, 0, stream>>>(src, dst, Cout, Kraw, Kp, n);
    };
    auto launch_conv3 = [&](const bf16* X, const bf16* Wt, float* Y,
                            int Cin, int H, int W, int Cout, int Ho, int Wo,
                            int stride, int Kp) {
        dim3 g(cdivi(Ho * Wo, 256), cdivi(Cout, 128), B);
        k_conv_wmma<3,3,1><<<g, 256, 0, stream>>>(X, Wt, Y, Cin, H, W, Cout, Ho, Wo, stride, Kp);
    };
    auto launch_conv1 = [&](const bf16* X, const bf16* Wt, float* Y,
                            int Cin, int H, int W, int Cout, int Ho, int Wo,
                            int stride, int Kp) {
        dim3 g(cdivi(Ho * Wo, 256), cdivi(Cout, 128), B);
        k_conv_wmma<1,1,0><<<g, 256, 0, stream>>>(X, Wt, Y, Cin, H, W, Cout, Ho, Wo, stride, Kp);
    };

    // ---- stem: conv0 -> inorm -> prelu ----
    {
        int n = B * 3 * 256 * 256;
        k_cvt_bf16<<<cdivi(n, 256), 256, 0, stream>>>(x_in, BFB, n);
        int Kp = kpad32(3 * 9);   // 27 -> 32
        cvtw(conv0w, W1B, 64, 27, Kp);
        launch_conv3(BFB, W1B, T1, 3, 256, 256, 64, 256, 256, 1, Kp);
        k_inorm_stats<<<B * 64, 256, 0, stream>>>(T1, MEAN, RSTD, 256 * 256);
        k_inorm_prelu<<<B * 64, 256, 0, stream>>>(T1, MEAN, RSTD, a0, X0, 64, 256 * 256);
    }

    float* cur = X0;
    float* nxt = X1;
    int H = 256, W = 256;

    for (int i = 0; i < 24; ++i) {
        const int inc = cfg[i].inc, dep = cfg[i].dep, st = cfg[i].st;
        const int Ho = H / st, Wo = W / st;
        const int HW = H * W, HoWo = Ho * Wo;

        // ---- shortcut ----
        const float* scPtr;
        if (inc == dep && st == 1) {
            scPtr = cur;
        } else if (inc == dep) {
            int n = B * dep * HoWo;
            k_slice2<<<cdivi(n, 256), 256, 0, stream>>>(cur, SC, H, W, Ho, Wo, n);
            scPtr = SC;
        } else {
            int n = B * inc * HW;
            k_cvt_bf16<<<cdivi(n, 256), 256, 0, stream>>>(cur, BFB, n);
            int Kp = kpad32(inc);
            cvtw(scw[i], WSB, dep, inc, Kp);
            launch_conv1(BFB, WSB, SC, inc, H, W, dep, Ho, Wo, st, Kp);
            k_inorm_stats<<<B * dep, 256, 0, stream>>>(SC, MEAN2, RSTD2, HoWo);
            k_inorm_f32<<<B * dep, 256, 0, stream>>>(SC, MEAN2, RSTD2, SC, HoWo);
            scPtr = SC;
        }

        // ---- residual path: inorm -> conv1 -> prelu -> conv2 -> inorm -> SE -> +sc ----
        k_inorm_stats<<<B * inc, 256, 0, stream>>>(cur, MEAN, RSTD, HW);
        k_inorm_bf16<<<B * inc, 256, 0, stream>>>(cur, MEAN, RSTD, BFA, HW);

        int Kp1 = kpad32(inc * 9);
        cvtw(w1[i], W1B, dep, inc * 9, Kp1);
        launch_conv3(BFA, W1B, T1, inc, H, W, dep, H, W, 1, Kp1);

        k_prelu_bf16<<<B * dep, 256, 0, stream>>>(T1, ap[i], BFA, dep, HW);

        int Kp2 = kpad32(dep * 9);
        cvtw(w2[i], W2B, dep, dep * 9, Kp2);
        launch_conv3(BFA, W2B, T2, dep, H, W, dep, Ho, Wo, st, Kp2);

        k_inorm_stats<<<B * dep, 256, 0, stream>>>(T2, MEAN, RSTD, HoWo);
        k_se_gate<<<1, 256, 0, stream>>>(sb1[i], sw2[i], sb2[i], G, dep, dep / 16);
        k_combine<<<B * dep, 256, 0, stream>>>(T2, MEAN, RSTD, G, scPtr, nxt, dep, HoWo);

        // ---- taps -> segmap pooling straight into d_out ----
        if (i == 6)
            k_pool_seg<<<dim3(S, B), 256, 0, stream>>>(nxt, segmap, codes, 256, 64, 64, S, 0,   Ctot);
        else if (i == 20)
            k_pool_seg<<<dim3(S, B), 256, 0, stream>>>(nxt, segmap, codes, 512, 32, 32, S, 256, Ctot);
        else if (i == 23)
            k_pool_seg<<<dim3(S, B), 256, 0, stream>>>(nxt, segmap, codes, 512, 16, 16, S, 768, Ctot);

        float* t = cur; cur = nxt; nxt = t;
        H = Ho; W = Wo;
    }

    // ---- second output: zeros_like(x) = (2,512,16,16) ----
    {
        int n = B * 512 * 16 * 16;
        k_zero<<<cdivi(n, 256), 256, 0, stream>>>(ztail, n);
    }
}
